// _GAT_59390807769119
// MI455X (gfx1250) — compile-verified, matrix operand order checked by reading, not run on hardware
//
#include <hip/hip_runtime.h>
#include <hip/hip_bf16.h>
#include <math.h>

typedef __attribute__((ext_vector_type(16))) _Float16 v16h;
typedef __attribute__((ext_vector_type(8)))  float    v8f;

#define HD        128   // H * DH
#define NHEAD     4
#define DHEAD     32
#define FIN       256
#define NEG_SLOPE 0.2f

// ---------- helpers ----------

__device__ __forceinline__ float atomic_max_f32(float* addr, float v) {
  // IEEE trick: int compare == float compare for v>=0; reversed unsigned for v<0
  if (v >= 0.0f)
    return __int_as_float(atomicMax((int*)addr, __float_as_int(v)));
  else
    return __uint_as_float(atomicMin((unsigned int*)addr, __float_as_uint(v)));
}

__device__ __forceinline__ void edge_sd(const int* __restrict__ ei, int E, int e,
                                        int& s, int& d) {
  if (e < E) { s = ei[e]; d = ei[E + e]; }
  else       { s = e - E; d = s; }          // appended self-loops
}

// fragment-order k index: element i of a lane's 16 halves, hi = lane>>4
__device__ __forceinline__ int frag_k(int kb, int hi, int i) {
  return kb * 32 + (i < 8 ? hi * 8 + i : 16 + hi * 8 + (i - 8));
}

// ---------- pack kernels: f32 row-major -> f16 WMMA fragment order ----------
// Layout: P[(((tile * KB) + kb) * 32 + lane) * 16 + i], 32B contiguous per lane.

// A (M x K): tile = M/16 row tile, lane&15 = row-in-tile
__global__ void k_pack_a(const float* __restrict__ X, _Float16* __restrict__ Ap,
                         int M, int K) {
  int tid = blockIdx.x * blockDim.x + threadIdx.x;
  if (tid >= M * K) return;
  const int KB = K >> 5;
  int o = tid;
  int i  = o & 15; o >>= 4;
  int l  = o & 31; o >>= 5;
  int kb = o % KB;
  int mt = o / KB;
  int row = mt * 16 + (l & 15);
  int k   = frag_k(kb, l >> 4, i);
  Ap[tid] = (_Float16)X[(size_t)row * K + k];
}

// B (K x 128): tile = column tile (8 of them), lane&15 = col-in-tile
__global__ void k_pack_b(const float* __restrict__ W, _Float16* __restrict__ Bp, int K) {
  int tid = blockIdx.x * blockDim.x + threadIdx.x;
  if (tid >= K * HD) return;
  const int KB = K >> 5;
  int o = tid;
  int i  = o & 15; o >>= 4;
  int l  = o & 31; o >>= 5;
  int kb = o % KB;
  int t  = o / KB;
  int col = t * 16 + (l & 15);
  int k   = frag_k(kb, l >> 4, i);
  Bp[tid] = (_Float16)W[k * HD + col];
}

// ---------- WMMA GEMM: C[M,128] = A[M,K] * B[K,128], fragment-packed f16 ----------
// One block = 16 rows; 8 waves = 8 column tiles of 16. Requires M % 16 == 0.
template <int K>
__global__ __launch_bounds__(256)
void k_gemm_f16_wmma(const _Float16* __restrict__ Ap, const _Float16* __restrict__ Bp,
                     float* __restrict__ C) {
  constexpr int KB = K / 32;
  const int lane = threadIdx.x & 31;
  const int wave = threadIdx.x >> 5;      // column tile
  const int lo   = lane & 15;
  const int hi   = lane >> 4;
  const int m0   = blockIdx.x * 16;

  const _Float16* __restrict__ abase = Ap + ((size_t)blockIdx.x * KB * 32 + lane) * 16;
  const _Float16* __restrict__ bbase = Bp + ((size_t)wave       * KB * 32 + lane) * 16;

  v8f c = {};
#pragma unroll
  for (int kb = 0; kb < KB; ++kb) {
    v16h a = *(const v16h*)(abase + (size_t)kb * 512);   // 2x global_load_b128
    v16h b = *(const v16h*)(bbase + (size_t)kb * 512);   // 2x global_load_b128
    c = __builtin_amdgcn_wmma_f32_16x16x32_f16(false, a, false, b,
                                               (short)0, c, false, false);
  }

  // C/D layout: VGPR r -> row m0 + hi*8 + r, column = wave*16 + lo
  float* __restrict__ cp = C + (size_t)(m0 + hi * 8) * HD + wave * 16 + lo;
#pragma unroll
  for (int r = 0; r < 8; ++r) cp[(size_t)r * HD] = c[r];
}

// ---------- elementwise ----------

__global__ void k_fill_f32(float* __restrict__ p, float v, int n) {
  int i = blockIdx.x * blockDim.x + threadIdx.x;
  if (i < n) p[i] = v;
}

// out[i] += bias[i % 128]; optionally elu_out[i] = elu(out[i]) (f32)
__global__ void k_bias_elu(float* __restrict__ out, const float* __restrict__ bias,
                           float* __restrict__ elu_out, int n) {
  int i = blockIdx.x * blockDim.x + threadIdx.x;
  if (i >= n) return;
  float v = out[i] + bias[i & (HD - 1)];
  out[i] = v;
  if (elu_out) elu_out[i] = v > 0.0f ? v : (expf(v) - 1.0f);
}

// ---------- attention coefficients per node/head ----------
__global__ void k_alpha(const float* __restrict__ h, const float* __restrict__ a_src,
                        const float* __restrict__ a_dst,
                        float* __restrict__ as, float* __restrict__ ad, int n) {
  int tid = blockIdx.x * blockDim.x + threadIdx.x;
  if (tid >= n * NHEAD) return;
  int node = tid >> 2, hd = tid & 3;
  const float* hp = h + (size_t)node * HD + hd * DHEAD;
  const float* sp = a_src + hd * DHEAD;
  const float* dp = a_dst + hd * DHEAD;
  float ss = 0.0f, dd = 0.0f;
#pragma unroll
  for (int i = 0; i < DHEAD; ++i) { float v = hp[i]; ss += v * sp[i]; dd += v * dp[i]; }
  as[tid] = ss;
  ad[tid] = dd;
}

// ---------- segment softmax over destination nodes ----------
__global__ void k_edge_max(const int* __restrict__ ei, int E, int Etot,
                           const float* __restrict__ as, const float* __restrict__ ad,
                           float* __restrict__ emax) {
  int tid = blockIdx.x * blockDim.x + threadIdx.x;
  if (tid >= Etot * NHEAD) return;
  int e = tid >> 2, hd = tid & 3, s, d;
  edge_sd(ei, E, e, s, d);
  float v = as[s * NHEAD + hd] + ad[d * NHEAD + hd];
  v = v > 0.0f ? v : NEG_SLOPE * v;
  atomic_max_f32(&emax[d * NHEAD + hd], v);
}

// also caches the unnormalized exp per (edge, head) for the aggregation pass
__global__ void k_edge_sum(const int* __restrict__ ei, int E, int Etot,
                           const float* __restrict__ as, const float* __restrict__ ad,
                           const float* __restrict__ emax,
                           float* __restrict__ denom, float* __restrict__ exb) {
  int tid = blockIdx.x * blockDim.x + threadIdx.x;
  if (tid >= Etot * NHEAD) return;
  int e = tid >> 2, hd = tid & 3, s, d;
  edge_sd(ei, E, e, s, d);
  float v = as[s * NHEAD + hd] + ad[d * NHEAD + hd];
  v = v > 0.0f ? v : NEG_SLOPE * v;
  float ex = expf(v - emax[d * NHEAD + hd]);
  exb[tid] = ex;
  atomicAdd(&denom[d * NHEAD + hd], ex);
}

// one thread per (edge, feature); 32 consecutive threads read consecutive h[src] elems
__global__ void k_edge_agg(const int* __restrict__ ei, int E, int Etot,
                           const float* __restrict__ exb, const float* __restrict__ denom,
                           const float* __restrict__ h, float* __restrict__ out) {
  int tid = blockIdx.x * blockDim.x + threadIdx.x;
  if (tid >= Etot * HD) return;
  int e = tid >> 7, r = tid & (HD - 1), hd = r >> 5, s, d;
  edge_sd(ei, E, e, s, d);
  float alpha = exb[e * NHEAD + hd] / denom[d * NHEAD + hd];
  atomicAdd(&out[(size_t)d * HD + r], h[(size_t)s * HD + r] * alpha);
}

// ---------- host orchestration ----------

static inline int gr(long long n, int b) { return (int)((n + b - 1) / b); }

extern "C" void kernel_launch(void* const* d_in, const int* in_sizes, int n_in,
                              void* d_out, int out_size, void* d_ws, size_t ws_size,
                              hipStream_t stream) {
  const float* x   = (const float*)d_in[0];
  const int*   ei  = (const int*)  d_in[1];
  const float* W1  = (const float*)d_in[2];
  const float* as1 = (const float*)d_in[3];
  const float* ad1 = (const float*)d_in[4];
  const float* b1  = (const float*)d_in[5];
  const float* W2  = (const float*)d_in[6];
  const float* as2 = (const float*)d_in[7];
  const float* ad2 = (const float*)d_in[8];
  const float* b2  = (const float*)d_in[9];

  const int N    = in_sizes[0] / FIN;
  const int E    = in_sizes[1] / 2;
  const int Etot = E + N;

  float* out_x = (float*)d_out;
  float* out1  = out_x + (size_t)N * FIN;
  float* out2  = out1 + (size_t)N * HD;

  // workspace layout (all offsets 32B aligned)
  char* ws = (char*)d_ws;
  float*    hbuf  = (float*)ws;        ws += (size_t)N * HD * sizeof(float);
  float*    asb   = (float*)ws;        ws += (size_t)N * NHEAD * sizeof(float);
  float*    adb   = (float*)ws;        ws += (size_t)N * NHEAD * sizeof(float);
  float*    emax  = (float*)ws;        ws += (size_t)N * NHEAD * sizeof(float);
  float*    denom = (float*)ws;        ws += (size_t)N * NHEAD * sizeof(float);
  float*    exb   = (float*)ws;        ws += (size_t)Etot * NHEAD * sizeof(float);
  _Float16* Apack = (_Float16*)ws;     ws += (size_t)N * FIN * sizeof(_Float16);
  _Float16* Bpack = (_Float16*)ws;     /* FIN*HD halves */

  const int B = 256;

  // output[0]: passthrough x
  hipMemcpyAsync(d_out, x, (size_t)N * FIN * sizeof(float),
                 hipMemcpyDeviceToDevice, stream);

  const float* asv[2]  = {as1, as2};
  const float* adv[2]  = {ad1, ad2};
  const float* bias[2] = {b1, b2};
  float*       outs[2] = {out1, out2};

  for (int layer = 0; layer < 2; ++layer) {
    const int K = layer == 0 ? FIN : HD;

    // pack operands into WMMA fragment order (fuses the f32->f16 cast)
    k_pack_a<<<gr((long long)N * K, B), B, 0, stream>>>(layer == 0 ? x : hbuf,
                                                        Apack, N, K);
    k_pack_b<<<gr((long long)K * HD, B), B, 0, stream>>>(layer == 0 ? W1 : W2, Bpack, K);

    // h = A @ W (WMMA f16 -> f32), fully unrolled K loop
    if (layer == 0) k_gemm_f16_wmma<FIN><<<N / 16, B, 0, stream>>>(Apack, Bpack, hbuf);
    else            k_gemm_f16_wmma<HD> <<<N / 16, B, 0, stream>>>(Apack, Bpack, hbuf);

    // per-node attention logits
    k_alpha<<<gr((long long)N * NHEAD, B), B, 0, stream>>>(hbuf, asv[layer], adv[layer],
                                                           asb, adb, N);

    // init softmax stats + output accumulator
    k_fill_f32<<<gr((long long)N * NHEAD, B), B, 0, stream>>>(emax, -INFINITY, N * NHEAD);
    k_fill_f32<<<gr((long long)N * NHEAD, B), B, 0, stream>>>(denom, 0.0f, N * NHEAD);
    k_fill_f32<<<gr((long long)N * HD, B), B, 0, stream>>>(outs[layer], 0.0f, N * HD);

    // segment softmax + weighted aggregation
    k_edge_max<<<gr((long long)Etot * NHEAD, B), B, 0, stream>>>(ei, E, Etot, asb, adb, emax);
    k_edge_sum<<<gr((long long)Etot * NHEAD, B), B, 0, stream>>>(ei, E, Etot, asb, adb,
                                                                 emax, denom, exb);
    k_edge_agg<<<gr((long long)Etot * HD, B), B, 0, stream>>>(ei, E, Etot, exb, denom,
                                                              hbuf, outs[layer]);

    // bias; between layers also ELU (f32) into hbuf for the next pack_a
    k_bias_elu<<<gr((long long)N * HD, B), B, 0, stream>>>(outs[layer], bias[layer],
                                                           layer == 0 ? hbuf : (float*)nullptr,
                                                           N * HD);
  }
}
